// LSTM_5566277616549
// MI455X (gfx1250) — matrix operand order repaired and checked by reading, not compiled, not run on hardware
//
#include <hip/hip_runtime.h>

// ---- CDNA5 vector types for WMMA ----
typedef __bf16 v16bf __attribute__((ext_vector_type(16)));
typedef __bf16 v8bf  __attribute__((ext_vector_type(8)));
typedef float  v8f   __attribute__((ext_vector_type(8)));

#define F_IN  256
#define HDIM  512
#define SEQ   1024
#define NB    64
#define KTOT  768          // F_IN + HDIM
#define KP    784          // padded K stride (bank-spread: 784*2B = 49*32B)
#define ROWS  16           // batch rows per workgroup (one WMMA M-tile)
#define COLS  32           // h-columns per workgroup
#define NWGC  16           // workgroups per cluster (column split)

__device__ __forceinline__ unsigned short f2bf(float f) {
  unsigned u = __float_as_uint(f);
  u += 0x7FFF + ((u >> 16) & 1);   // round-to-nearest-even
  return (unsigned short)(u >> 16);
}
// Branch-free activations: v_exp_f32 + fast divide, no EXEC-divergent slow path.
__device__ __forceinline__ float fast_sigmoid(float x) {
  float xc = fminf(fmaxf(x, -30.0f), 30.0f);
  return __fdividef(1.0f, 1.0f + __expf(-xc));
}
__device__ __forceinline__ float fast_tanh(float x) {
  float xc = fminf(fmaxf(x, -15.0f), 15.0f);
  float e = __expf(2.0f * xc);
  return __fdividef(e - 1.0f, e + 1.0f);
}

union Frag { v16bf v; v8bf h[2]; };

__global__ __launch_bounds__(128, 1)
void lstm_persistent(const float* __restrict__ x,
                     const float* __restrict__ Wx0, const float* __restrict__ Wh0, const float* __restrict__ b0,
                     const float* __restrict__ Wx1, const float* __restrict__ Wh1, const float* __restrict__ b1,
                     const float* __restrict__ Wx2, const float* __restrict__ Wh2, const float* __restrict__ b2,
                     const float* __restrict__ Wx3, const float* __restrict__ Wh3, const float* __restrict__ b3,
                     unsigned short* __restrict__ hx,   // [64][512] bf16 h-exchange (in d_ws)
                     float* __restrict__ out)           // h[64][512] then c[64][512]
{
  extern __shared__ char smem[];
  unsigned short* Wl   = (unsigned short*)smem;                       // [128 cols][KP] bf16  (~196 KB)
  unsigned short* Al   = (unsigned short*)(smem + 128 * KP * 2);      // [16 rows][KP] bf16   (~25 KB)
  float*          zbuf = (float*)(smem + 128 * KP * 2 + 16 * KP * 2); // [8 tiles][16][16]    (8 KB)
  float*          cbuf = (float*)((char*)zbuf + 8 * 256 * 4);         // [16][32]             (2 KB)

  const int tid  = threadIdx.x;
  const int bg   = blockIdx.x / NWGC;   // batch group 0..3  (one cluster each)
  const int cg   = blockIdx.x % NWGC;   // column group 0..15 (one WGP each)
  const int hc0  = cg * COLS;

  const int lane = tid & 31;
  const int wv   = tid >> 5;            // wave id == gate id (i,f,g,o)
  const int sel  = lane >> 4;           // lane half
  const int nn   = lane & 15;

  const float* WxA[4] = {Wx0, Wx1, Wx2, Wx3};
  const float* WhA[4] = {Wh0, Wh1, Wh2, Wh3};
  const float* bA[4]  = {b0, b1, b2, b3};

  // ---- one-time: load this WGP's [768 x 128] weight slice into LDS as bf16,
  //      column-major with K contiguous (B-fragment friendly) ----
  {
    const int lc  = tid;                 // local column 0..127
    const int t   = lc >> 4;             // N-tile 0..7
    const int g   = t >> 1;              // gate
    const int sub = t & 1;
    const int col = hc0 + sub * 16 + (lc & 15);
    const float* wx = WxA[g];
    const float* wh = WhA[g];
    for (int k = 0; k < F_IN; ++k)
      Wl[lc * KP + k] = f2bf(wx[k * HDIM + col]);
    for (int k = 0; k < HDIM; ++k)
      Wl[lc * KP + F_IN + k] = f2bf(wh[k * HDIM + col]);
  }
  for (int i = tid; i < ROWS * COLS; i += 128) cbuf[i] = 0.0f;
  for (int i = tid; i < ROWS * HDIM; i += 128)               // h0 = 0 in A-buffer
    Al[(i / HDIM) * KP + F_IN + (i % HDIM)] = 0;

  // ---- prologue: stage x_0 into A[k = 0..255] ----
  const int xrow = (tid + 0 * 128) >> 6; // row pattern reused below
  {
    #pragma unroll
    for (int j = 0; j < 8; ++j) {
      const int idx4 = tid + j * 128;
      const int row  = idx4 >> 6;
      const int f    = (idx4 & 63) << 2;
      const float4 xv = *(const float4*)(x + ((unsigned long long)(bg * ROWS + row) * SEQ + 0) * F_IN + f);
      uint2 p;
      p.x = (unsigned)f2bf(xv.x) | ((unsigned)f2bf(xv.y) << 16);
      p.y = (unsigned)f2bf(xv.z) | ((unsigned)f2bf(xv.w) << 16);
      *(uint2*)(&Al[row * KP + f]) = p;
    }
  }
  (void)xrow;
  __syncthreads();

  const float bias0 = bA[wv][hc0 + 0 * 16 + nn];   // tile 2*wv
  const float bias1 = bA[wv][hc0 + 1 * 16 + nn];   // tile 2*wv+1

  const int lc0 = (wv * 2 + 0) * 16 + nn;
  const int lc1 = (wv * 2 + 1) * 16 + nn;

  for (int s = 0; s < SEQ; ++s) {
    // ---- z = [x_t, h] * [Wx; Wh] : per wave, 2 N-tiles, K = 768 ----
    v8f acc0, acc1;
    #pragma unroll
    for (int r = 0; r < 8; ++r) { acc0[r] = bias0; acc1[r] = bias1; }

    #pragma unroll 4
    for (int ks = 0; ks < KTOT / 32; ++ks) {
      const int k0 = ks * 32;
      Frag a, br0, br1;
      const unsigned short* ap = &Al[nn * KP + k0 + sel * 8];
      a.h[0] = *(const v8bf*)(ap);          // K  k0+sel*8      .. +7
      a.h[1] = *(const v8bf*)(ap + 16);     // K  k0+16+sel*8   .. +7
      const unsigned short* bp0 = &Wl[lc0 * KP + k0 + sel * 8];
      br0.h[0] = *(const v8bf*)(bp0);
      br0.h[1] = *(const v8bf*)(bp0 + 16);
      const unsigned short* bp1 = &Wl[lc1 * KP + k0 + sel * 8];
      br1.h[0] = *(const v8bf*)(bp1);
      br1.h[1] = *(const v8bf*)(bp1 + 16);
      acc0 = __builtin_amdgcn_wmma_f32_16x16x32_bf16(false, a.v, false, br0.v, (short)0, acc0, false, false);
      acc1 = __builtin_amdgcn_wmma_f32_16x16x32_bf16(false, a.v, false, br1.v, (short)0, acc1, false, false);
    }

    // ---- spill z tiles to LDS so gates can be recombined per h-column ----
    #pragma unroll
    for (int r = 0; r < 8; ++r) {
      const int row = r + sel * 8;
      zbuf[(wv * 2 + 0) * 256 + row * 16 + nn] = acc0[r];
      zbuf[(wv * 2 + 1) * 256 + row * 16 + nn] = acc1[r];
    }
    __syncthreads();

    // ---- issue x_{s+1} global loads now; latency hides under gate math ----
    float4 xnext[8];
    const bool has_next = (s < SEQ - 1);
    if (has_next) {
      #pragma unroll
      for (int j = 0; j < 8; ++j) {
        const int idx4 = tid + j * 128;
        const int row  = idx4 >> 6;
        const int f    = (idx4 & 63) << 2;
        xnext[j] = *(const float4*)(x + ((unsigned long long)(bg * ROWS + row) * SEQ + (s + 1)) * F_IN + f);
      }
      if (s < SEQ - 2) {   // warm x_{s+2}: one 128B line per thread = full 16KB tile
        const int row = tid >> 3;
        const int f   = (tid & 7) << 5;
        __builtin_prefetch(x + ((unsigned long long)(bg * ROWS + row) * SEQ + (s + 2)) * F_IN + f, 0, 1);
      }
    }

    // ---- elementwise cell update: 512 elements / 128 threads ----
    #pragma unroll
    for (int e = 0; e < 4; ++e) {
      const int idx = tid + e * 128;             // 0..511
      const int row = idx >> 5, col = idx & 31;
      const int zo_ = (col >> 4) * 256 + row * 16 + (col & 15);
      const float zi = zbuf[0 * 512 + zo_];
      const float zf = zbuf[1 * 512 + zo_];
      const float zg = zbuf[2 * 512 + zo_];
      const float zq = zbuf[3 * 512 + zo_];
      const float ig = fast_sigmoid(zi), fg = fast_sigmoid(zf);
      const float gg = fast_tanh(zg),    og = fast_sigmoid(zq);
      const float c  = fg * cbuf[idx] + ig * gg;
      const float h  = og * fast_tanh(c);
      cbuf[idx] = c;
      const unsigned long long o = (unsigned long long)(bg * ROWS + row) * HDIM + hc0 + col;
      hx[o] = f2bf(h);
      if (s == SEQ - 1) {                        // final (h, c) in f32
        out[o] = h;
        out[(unsigned long long)NB * HDIM + o] = c;
      }
    }
    __threadfence();       // make hx stores visible across the cluster
    __syncthreads();

    if (has_next) {
      // ---- split cluster barrier: signal, overlap x staging, then wait ----
      asm volatile("s_barrier_signal -3" ::: "memory");

      // convert + stage x_{s+1} into A[k = 0..255] inside the barrier window
      #pragma unroll
      for (int j = 0; j < 8; ++j) {
        const int idx4 = tid + j * 128;
        const int row  = idx4 >> 6;
        const int f    = (idx4 & 63) << 2;
        uint2 p;
        p.x = (unsigned)f2bf(xnext[j].x) | ((unsigned)f2bf(xnext[j].y) << 16);
        p.y = (unsigned)f2bf(xnext[j].z) | ((unsigned)f2bf(xnext[j].w) << 16);
        *(uint2*)(&Al[row * KP + f]) = p;
      }

      asm volatile("s_barrier_wait -3" ::: "memory");

      // ---- async copy full 16x512 bf16 h block into A[k = 256..767] ----
      #pragma unroll
      for (int j = 0; j < 8; ++j) {
        const int ch  = tid + j * 128;           // 1024 x 16B chunks
        const int row = ch >> 6;
        const int off = (ch & 63) * 16;
        const unsigned long long ga =
            (unsigned long long)(const void*)hx +
            (unsigned long long)(bg * ROWS + row) * HDIM * 2 + (unsigned)off;
        const unsigned lds =
            (unsigned)(unsigned long long)(const void*)&Al[row * KP + F_IN] + (unsigned)off;
        asm volatile("global_load_async_to_lds_b128 %0, %1, off"
                     :: "v"(lds), "v"(ga) : "memory");
      }
      asm volatile("s_wait_asynccnt 0" ::: "memory");
      __syncthreads();
    }
  }
}

extern "C" void kernel_launch(void* const* d_in, const int* in_sizes, int n_in,
                              void* d_out, int out_size, void* d_ws, size_t ws_size,
                              hipStream_t stream) {
  (void)in_sizes; (void)n_in; (void)out_size; (void)ws_size;
  const float* x   = (const float*)d_in[0];
  const float* Wii = (const float*)d_in[1];
  const float* Wih = (const float*)d_in[2];
  const float* bi  = (const float*)d_in[3];
  const float* Wfi = (const float*)d_in[4];
  const float* Wfh = (const float*)d_in[5];
  const float* bf  = (const float*)d_in[6];
  const float* Wgi = (const float*)d_in[7];
  const float* Wgh = (const float*)d_in[8];
  const float* bg  = (const float*)d_in[9];
  const float* Woi = (const float*)d_in[10];
  const float* Woh = (const float*)d_in[11];
  const float* bo  = (const float*)d_in[12];

  unsigned short* hx = (unsigned short*)d_ws;    // 64*512 bf16 = 64 KB h-exchange
  float* out = (float*)d_out;

  const size_t shmem = (size_t)(128 * KP * 2)    // weights
                     + (size_t)(16 * KP * 2)     // A = [x_t, h]
                     + (size_t)(8 * 256 * 4)     // z tiles
                     + (size_t)(ROWS * COLS * 4);// c state   => 236,032 B < 320 KB

  // 4 batch-group clusters x 16 column-group WGPs, 128 threads (4 waves = 4 gates)
  lstm_persistent<<<dim3(NB / ROWS * NWGC), dim3(128), shmem, stream>>>(
      x, Wii, Wih, bi, Wfi, Wfh, bf, Wgi, Wgh, bg, Woi, Woh, bo, hx, out);
}
